// DeterministicDecoder_73564199846243
// MI455X (gfx1250) — matrix operand order repaired and checked by reading, not compiled, not run on hardware
//
#include <hip/hip_runtime.h>

typedef __bf16 bf16;
typedef __attribute__((ext_vector_type(16))) __bf16 v16bf;
typedef __attribute__((ext_vector_type(8)))  __bf16 v8bf;
typedef __attribute__((ext_vector_type(8)))  float  v8f;
typedef __attribute__((ext_vector_type(4)))  int    v4i;
typedef __attribute__((ext_vector_type(8)))  int    v8i;
typedef __attribute__((ext_vector_type(4)))  unsigned int v4u;

#define LN_EPS 1e-6f

#if defined(__AMDGCN__) && __has_builtin(__builtin_amdgcn_tensor_load_to_lds)
#define HAVE_TDM 1
#else
#define HAVE_TDM 0
#endif

// ---------------------------------------------------------------------------
// CDNA5 wait helpers
// ---------------------------------------------------------------------------
__device__ __forceinline__ void wait_async0() {
#if defined(__AMDGCN__)
#if __has_builtin(__builtin_amdgcn_s_wait_asynccnt)
  __builtin_amdgcn_s_wait_asynccnt(0);
#else
  asm volatile("s_wait_asynccnt 0x0" ::: "memory");
#endif
#endif
}

__device__ __forceinline__ void wait_tensor0() {
#if defined(__AMDGCN__)
#if __has_builtin(__builtin_amdgcn_s_wait_tensorcnt)
  __builtin_amdgcn_s_wait_tensorcnt(0);
#else
  asm volatile("s_wait_tensorcnt 0x0" ::: "memory");
#endif
#endif
}

__device__ __forceinline__ void wait_stage() {
#if HAVE_TDM
  wait_tensor0();
#else
  wait_async0();
#endif
}

// ---------------------------------------------------------------------------
// async global->LDS b128 copy (fallback staging path, ASYNCcnt-tracked)
// ---------------------------------------------------------------------------
__device__ __forceinline__ void async_copy_b128(void* lds, const void* g) {
#if defined(__AMDGCN__) && __has_builtin(__builtin_amdgcn_global_load_async_to_lds_b128)
  typedef v4i __attribute__((address_space(1)))* gvp;
  typedef v4i __attribute__((address_space(3)))* lvp;
  __builtin_amdgcn_global_load_async_to_lds_b128((gvp)g, (lvp)lds, 0, 0);
#else
  *(v8bf*)lds = *(const v8bf*)g;  // synchronous fallback
#endif
}

// ---------------------------------------------------------------------------
// TDM: stage a [rows=128, k=32] bf16 tile (row stride Kd elems) into LDS with
// hardware padding: 64B row + 16B pad -> LDS row stride 80B (AST=40 elems).
// Issued by one wave per workgroup; tracked on TENSORcnt.
// D# bitfields per CDNA5 ISA ch.8 (group0 128b, group1 256b, groups 2/3 zero).
// ---------------------------------------------------------------------------
#if HAVE_TDM
__device__ __forceinline__ void tdm_stage_A(const void* g, unsigned ldsOff,
                                            int Kd, int Mr) {
  unsigned long long ga = (unsigned long long)(uintptr_t)g;
  v4u g0;
  g0[0] = 1u;                                     // count=1, user desc, no gather
  g0[1] = ldsOff;                                 // lds_addr (bytes)
  g0[2] = (unsigned)(ga & 0xFFFFFFFFu);           // global_addr[31:0]
  g0[3] = (unsigned)((ga >> 32) & 0x1FFFFFFu)     // global_addr[56:32]
          | (2u << 30);                           // type=2 ("image")
  const unsigned td0 = (unsigned)Kd;              // tensor_dim0 (elems)
  const unsigned td1 = (unsigned)Mr;              // tensor_dim1 (rows)
  const unsigned long long st0 = (unsigned long long)(unsigned)Kd;  // dim0 stride
  v8i g1;
  g1[0] = (int)((1u << 16)                        // data_size = 2 bytes
              | (1u << 20)                        // pad_enable
              | (3u << 22)                        // pad_interval: 16 DWORDs (64B)
              | (3u << 25));                      // pad_amount: 4 DWORDs (16B)
  g1[1] = (int)((td0 & 0xFFFFu) << 16);           // [47:32]=0, [63:48]=td0.lo
  g1[2] = (int)((td0 >> 16) | ((td1 & 0xFFFFu) << 16));
  g1[3] = (int)((td1 >> 16) | (32u << 16));       // tile_dim0 = 32
  g1[4] = (int)128u;                              // tile_dim1 = 128, tile_dim2 = 0
  g1[5] = (int)(st0 & 0xFFFFFFFFu);
  g1[6] = (int)((st0 >> 32) & 0xFFFFu);           // dim1_stride unused (2D)
  g1[7] = 0;
  v4i z4 = {0, 0, 0, 0};
  v8i z8 = {0, 0, 0, 0, 0, 0, 0, 0};
  // 6-arg form (this toolchain): (g0, g1, g2, g3, extra_group, cpol)
  __builtin_amdgcn_tensor_load_to_lds(g0, g1, z4, z4, z8, 0);
}
#endif

// ---------------------------------------------------------------------------
// elementwise f32 -> bf16 cast
// ---------------------------------------------------------------------------
__global__ __launch_bounds__(256)
void cast_f32_bf16_kernel(const float* __restrict__ in, bf16* __restrict__ out, size_t n) {
  size_t i = (size_t)blockIdx.x * blockDim.x + threadIdx.x;
  if (i < n) out[i] = (bf16)in[i];
}

// f32 [rows, cin] -> bf16 [rows, cout] zero-padded cast
__global__ __launch_bounds__(256)
void cast_pad_bf16_kernel(const float* __restrict__ in, bf16* __restrict__ out,
                          int rows, int cin, int cout) {
  size_t idx = (size_t)blockIdx.x * blockDim.x + threadIdx.x;
  if (idx >= (size_t)rows * cout) return;
  int r = (int)(idx / cout), c = (int)(idx % cout);
  out[idx] = (c < cin) ? (bf16)in[(size_t)r * cin + c] : (bf16)0.f;
}

// ---------------------------------------------------------------------------
// bf16 transpose (in [R, C] -> out [C, R]) via padded LDS tile
// ---------------------------------------------------------------------------
__global__ __launch_bounds__(256)
void transpose_bf16_kernel(const bf16* __restrict__ in, bf16* __restrict__ out,
                           int R, int Ccol) {
  __shared__ bf16 tile[32][33];
  int c0 = blockIdx.x * 32;
  int r0 = blockIdx.y * 32;
  int tx = threadIdx.x, ty = threadIdx.y;  // 32 x 8
#pragma unroll
  for (int i = 0; i < 4; ++i) {
    int r = r0 + ty + i * 8, c = c0 + tx;
    if (r < R && c < Ccol) tile[ty + i * 8][tx] = in[(size_t)r * Ccol + c];
  }
  __syncthreads();
#pragma unroll
  for (int i = 0; i < 4; ++i) {
    int r = c0 + ty + i * 8;
    int c = r0 + tx;
    if (r < Ccol && c < R) out[(size_t)r * R + c] = tile[tx][ty + i * 8];
  }
}

// ---------------------------------------------------------------------------
// Tiled bf16 GEMM via V_WMMA_F32_16X16X32_BF16, double-buffered LDS,
// TDM (tensor_load_to_lds) staging of the A tile.
//   out[Mr,NcOut] = act( alpha * A[Mr,Kd] @ B[Kd,NcPad] + bias + addend )
// Requirements: Mr % 128 == 0, NcPad % 128 == 0, Kd % 32 == 0.
// Block tile 128x128, 8 waves (4 along M x 2 along N), wave tile 32x64.
// act: 0 = none (store outF f32 and/or outB bf16)
//      1 = hash: prob = sigmoid(x); outF = code (= prob > 0.5), outF2 = prob
// ---------------------------------------------------------------------------
#define AST 40
#define BST 40

__global__ __launch_bounds__(256)
void gemm_bf16_wmma_kernel(const bf16* __restrict__ A, const bf16* __restrict__ Bm,
                           int Mr, int NcPad, int NcOut, int Kd, float alpha,
                           const float* __restrict__ bias,
                           const float* __restrict__ addend,
                           float* __restrict__ outF,
                           float* __restrict__ outF2,
                           bf16* __restrict__ outB,
                           int act) {
  __shared__ bf16 sA[2][128 * AST];
  __shared__ bf16 sB[2][128 * BST];

  const int tid  = threadIdx.x;
  const int lane = tid & 31;
  const int wave = tid >> 5;
  const int wm   = wave & 3;   // M direction, 32 rows each
  const int wn   = wave >> 2;  // N direction, 64 cols each
  const int lane15   = lane & 15;
  const int laneHalf = lane >> 4;

  const int rowBase = blockIdx.y * 128;
  const int colBase = blockIdx.x * 128;

  // staging coordinates
  const int ar   = tid >> 2;           // (fallback path) A rows ar and ar+64
  const int acol = (tid & 3) << 3;
  const int bk   = tid >> 4;           // B k rows bk and bk+16
  const int bn   = (tid & 15) << 3;    // 8 consecutive n

  v8f acc[2][4];
#pragma unroll
  for (int i = 0; i < 2; ++i)
#pragma unroll
    for (int j = 0; j < 4; ++j)
      acc[i][j] = (v8f){0.f, 0.f, 0.f, 0.f, 0.f, 0.f, 0.f, 0.f};

  const int nk = Kd >> 5;

  // ---- A tile staging (TDM preferred) ----
  auto stage_A = [&](int k0, int buf) {
#if HAVE_TDM
    if (wave == 0) {
      tdm_stage_A(A + (size_t)rowBase * Kd + k0,
                  (unsigned)(uintptr_t)(&sA[buf][0]), Kd, Mr);
    }
#else
    async_copy_b128(&sA[buf][ar * AST + acol],
                    A + (size_t)(rowBase + ar) * Kd + k0 + acol);
    async_copy_b128(&sA[buf][(ar + 64) * AST + acol],
                    A + (size_t)(rowBase + ar + 64) * Kd + k0 + acol);
#endif
  };

  // ---- prologue: stage tile 0 into buffer 0 ----
  {
    stage_A(0, 0);
    v8bf b0 = *(const v8bf*)(Bm + (size_t)bk * NcPad + colBase + bn);
    v8bf b1 = *(const v8bf*)(Bm + (size_t)(bk + 16) * NcPad + colBase + bn);
    bf16* d0 = &sB[0][bn * BST + bk];
    bf16* d1 = &sB[0][bn * BST + bk + 16];
#pragma unroll
    for (int i = 0; i < 8; ++i) { d0[i * BST] = b0[i]; d1[i * BST] = b1[i]; }
  }
  wait_stage();
  __syncthreads();

  for (int t = 0; t < nk; ++t) {
    const int buf = t & 1;
    const bf16* cA = sA[buf];
    const bf16* cB = sB[buf];
    const bool more = (t + 1 < nk);

    // issue next tile's loads first: TDM A -> LDS, B -> registers
    v8bf nb0 = {}, nb1 = {};
    if (more) {
      const int k0 = (t + 1) << 5;
      stage_A(k0, buf ^ 1);
      nb0 = *(const v8bf*)(Bm + (size_t)(k0 + bk) * NcPad + colBase + bn);
      nb1 = *(const v8bf*)(Bm + (size_t)(k0 + bk + 16) * NcPad + colBase + bn);
    }

    // ---- gather fragments from current buffers and issue WMMAs ----
    v16bf aF[2], bF[4];
#pragma unroll
    for (int i = 0; i < 2; ++i) {
      // A 16-bit layout: lanes 0-15 rows 0..15; K = kHalf..+7 and kHalf+16..+23
      const bf16* p = cA + (wm * 32 + i * 16 + lane15) * AST + laneHalf * 8;
      union { v16bf v; v8bf h[2]; } u;
      u.h[0] = *(const v8bf*)p;
      u.h[1] = *(const v8bf*)(p + 16);
      aF[i] = u.v;
    }
#pragma unroll
    for (int j = 0; j < 4; ++j) {
      // B 16-bit layout: lane col = lane15, K = kHalf..+7 and kHalf+16..+23
      const bf16* p = cB + (wn * 64 + j * 16 + lane15) * BST + laneHalf * 8;
      union { v16bf v; v8bf h[2]; } u;
      u.h[0] = *(const v8bf*)p;
      u.h[1] = *(const v8bf*)(p + 16);
      bF[j] = u.v;
    }
#pragma unroll
    for (int i = 0; i < 2; ++i)
#pragma unroll
      for (int j = 0; j < 4; ++j)
        acc[i][j] = __builtin_amdgcn_wmma_f32_16x16x32_bf16(
            false, aF[i], false, bF[j], (short)0, acc[i][j], false, false);

    // deferred transposed B stores (global load latency hidden under WMMAs)
    if (more) {
      bf16* d0 = &sB[buf ^ 1][bn * BST + bk];
      bf16* d1 = &sB[buf ^ 1][bn * BST + bk + 16];
#pragma unroll
      for (int i = 0; i < 8; ++i) { d0[i * BST] = nb0[i]; d1[i * BST] = nb1[i]; }
    }

    wait_stage();
    __syncthreads();
  }

  // ---- epilogue ----
#pragma unroll
  for (int j = 0; j < 4; ++j) {
    const int cidx = colBase + wn * 64 + j * 16 + lane15;
    if (cidx >= NcOut) continue;
    const float bv = bias ? bias[cidx] : 0.f;
#pragma unroll
    for (int i = 0; i < 2; ++i) {
#pragma unroll
      for (int e = 0; e < 8; ++e) {
        const int r = rowBase + wm * 32 + i * 16 + laneHalf * 8 + e;
        float x = acc[i][j][e] * alpha + bv;
        const size_t o = (size_t)r * NcOut + cidx;
        if (addend) x += addend[o];
        if (act == 1) {
          float p    = 1.f / (1.f + __expf(-x));
          float code = (p > 0.5f) ? 1.f : 0.f;
          outF[o]  = code;
          outF2[o] = p;
        } else {
          if (outF) outF[o] = x;
          if (outB) outB[o] = (bf16)x;
        }
      }
    }
  }
}

// ---------------------------------------------------------------------------
// In-place row softmax on bf16 matrix [rows, Ncols], Ncols % 256 == 0, <= 4096
// ---------------------------------------------------------------------------
__global__ __launch_bounds__(256)
void softmax_bf16_kernel(bf16* __restrict__ s, int Ncols) {
  __shared__ float red[256];
  const int tid = threadIdx.x;
  bf16* p = s + (size_t)blockIdx.x * Ncols;
  const int niter = Ncols >> 8;  // <= 16

  float v[16];
  float mx = -3.0e38f;
  for (int i = 0; i < niter; ++i) {
    v[i] = (float)p[i * 256 + tid];
    mx = fmaxf(mx, v[i]);
  }
  red[tid] = mx;
  __syncthreads();
  for (int s2 = 128; s2 > 0; s2 >>= 1) {
    if (tid < s2) red[tid] = fmaxf(red[tid], red[tid + s2]);
    __syncthreads();
  }
  mx = red[0];
  __syncthreads();

  float sum = 0.f;
  for (int i = 0; i < niter; ++i) {
    v[i] = __expf(v[i] - mx);
    sum += v[i];
  }
  red[tid] = sum;
  __syncthreads();
  for (int s2 = 128; s2 > 0; s2 >>= 1) {
    if (tid < s2) red[tid] += red[tid + s2];
    __syncthreads();
  }
  float inv = 1.f / red[0];
  for (int i = 0; i < niter; ++i) p[i * 256 + tid] = (bf16)(v[i] * inv);
}

// ---------------------------------------------------------------------------
// Row LayerNorm: x [rows, Mcols] f32 -> bf16, Mcols % 256 == 0, <= 1024
// ---------------------------------------------------------------------------
__global__ __launch_bounds__(256)
void layernorm_kernel(const float* __restrict__ x,
                      const float* __restrict__ gamma,
                      const float* __restrict__ beta,
                      bf16* __restrict__ out, int Mcols) {
  __shared__ float red[256];
  const int tid = threadIdx.x;
  const float* p = x + (size_t)blockIdx.x * Mcols;
  const int niter = Mcols >> 8;  // <= 4

  float v[4];
  float sum = 0.f, sumsq = 0.f;
  for (int i = 0; i < niter; ++i) {
    v[i] = p[i * 256 + tid];
    sum += v[i];
    sumsq += v[i] * v[i];
  }
  red[tid] = sum;
  __syncthreads();
  for (int s2 = 128; s2 > 0; s2 >>= 1) {
    if (tid < s2) red[tid] += red[tid + s2];
    __syncthreads();
  }
  float mu = red[0] / (float)Mcols;
  __syncthreads();
  red[tid] = sumsq;
  __syncthreads();
  for (int s2 = 128; s2 > 0; s2 >>= 1) {
    if (tid < s2) red[tid] += red[tid + s2];
    __syncthreads();
  }
  float var  = red[0] / (float)Mcols - mu * mu;
  float rstd = __frsqrt_rn(var + LN_EPS);
  bf16* o = out + (size_t)blockIdx.x * Mcols;
  for (int i = 0; i < niter; ++i) {
    int c = i * 256 + tid;
    o[c] = (bf16)((v[i] - mu) * rstd * gamma[c] + beta[c]);
  }
}

// ---------------------------------------------------------------------------
// Tiny classifier: out[b,k] = sum_c code[b,c]*Wc[c,k] + bc[k]   (C=64, K=100)
// ---------------------------------------------------------------------------
__global__ __launch_bounds__(256)
void cls_kernel(const float* __restrict__ code, const float* __restrict__ Wc,
                const float* __restrict__ bc, float* __restrict__ out,
                int Bn, int C, int K) {
  size_t idx = (size_t)blockIdx.x * blockDim.x + threadIdx.x;
  if (idx >= (size_t)Bn * K) return;
  int b = (int)(idx / K), k = (int)(idx % K);
  float s = bc[k];
  const float* cp = code + (size_t)b * C;
  for (int c = 0; c < C; ++c) s += cp[c] * Wc[(size_t)c * K + k];
  out[idx] = s;
}

// ---------------------------------------------------------------------------
// launch
// ---------------------------------------------------------------------------
extern "C" void kernel_launch(void* const* d_in, const int* in_sizes, int n_in,
                              void* d_out, int out_size, void* d_ws, size_t ws_size,
                              hipStream_t stream) {
  const int B = 8192, N = 4096, D = 2048, M = 1024, C = 64, K = 100;
  const int CPAD = 128;

  const float* feat     = (const float*)d_in[0];
  const float* emb      = (const float*)d_in[1];
  const float* W_k      = (const float*)d_in[2];
  const float* W_v      = (const float*)d_in[3];
  const float* W_q      = (const float*)d_in[4];
  const float* W_hash   = (const float*)d_in[5];
  const float* b_hash   = (const float*)d_in[6];
  const float* ln_gamma = (const float*)d_in[7];
  const float* ln_beta  = (const float*)d_in[8];
  const float* W_cls    = (const float*)d_in[9];
  const float* b_cls    = (const float*)d_in[10];

  float* out_code = (float*)d_out;            // [B,C]
  float* out_prob = out_code + (size_t)B * C; // [B,C]
  float* out_cls  = out_prob + (size_t)B * C; // [B,K]

  char* ws = (char*)d_ws;
  size_t off = 0;
  auto alloc = [&](size_t bytes) -> char* {
    char* p = ws + off;
    off += (bytes + 255) & ~(size_t)255;
    return p;
  };

  bf16*  emb_bf    = (bf16*)alloc((size_t)N * D * 2);
  bf16*  feat_bf   = (bf16*)alloc((size_t)B * D * 2);
  bf16*  Wk_bf     = (bf16*)alloc((size_t)D * M * 2);
  bf16*  Wv_bf     = (bf16*)alloc((size_t)D * M * 2);
  bf16*  Wq_bf     = (bf16*)alloc((size_t)D * M * 2);
  bf16*  Wh_pad    = (bf16*)alloc((size_t)M * CPAD * 2);
  bf16*  k_bf      = (bf16*)alloc((size_t)N * M * 2);
  bf16*  kT_bf     = (bf16*)alloc((size_t)M * N * 2);
  bf16*  v_bf      = (bf16*)alloc((size_t)N * M * 2);
  float* q_f32     = (float*)alloc((size_t)B * M * 4);
  bf16*  q_bf      = (bf16*)alloc((size_t)B * M * 2);
  bf16*  scores_bf = (bf16*)alloc((size_t)B * N * 2);
  float* att_f32   = (float*)alloc((size_t)B * M * 4);
  bf16*  normed_bf = (bf16*)alloc((size_t)B * M * 2);

  auto cast = [&](const float* src, bf16* dst, size_t n) {
    cast_f32_bf16_kernel<<<dim3((unsigned)((n + 255) / 256)), dim3(256), 0, stream>>>(src, dst, n);
  };
  cast(emb,  emb_bf,  (size_t)N * D);
  cast(feat, feat_bf, (size_t)B * D);
  cast(W_k,  Wk_bf,   (size_t)D * M);
  cast(W_v,  Wv_bf,   (size_t)D * M);
  cast(W_q,  Wq_bf,   (size_t)D * M);
  cast_pad_bf16_kernel<<<dim3((unsigned)(((size_t)M * CPAD + 255) / 256)), dim3(256), 0, stream>>>(
      W_hash, Wh_pad, M, C, CPAD);

  auto gemm = [&](const bf16* A, const bf16* Bmat, int Mr, int NcPad, int NcOut, int Kd,
                  float alpha, const float* bias, const float* addend,
                  float* oF, float* oF2, bf16* oB, int act) {
    dim3 grid((NcPad + 127) / 128, (Mr + 127) / 128);
    gemm_bf16_wmma_kernel<<<grid, dim3(256), 0, stream>>>(
        A, Bmat, Mr, NcPad, NcOut, Kd, alpha, bias, addend, oF, oF2, oB, act);
  };

  // k = emb @ W_k ; v = emb @ W_v ; q = feat @ W_q (f32 + bf16)
  gemm(emb_bf,  Wk_bf, N, M, M, D, 1.f, nullptr, nullptr, nullptr, nullptr, k_bf, 0);
  gemm(emb_bf,  Wv_bf, N, M, M, D, 1.f, nullptr, nullptr, nullptr, nullptr, v_bf, 0);
  gemm(feat_bf, Wq_bf, B, M, M, D, 1.f, nullptr, nullptr, q_f32,  nullptr, q_bf, 0);

  // kT for scores GEMM
  transpose_bf16_kernel<<<dim3(M / 32, N / 32), dim3(32, 8), 0, stream>>>(k_bf, kT_bf, N, M);

  // scores = (q @ kT) / sqrt(M) -> bf16, softmax in place
  gemm(q_bf, kT_bf, B, N, N, M, 0.03125f, nullptr, nullptr, nullptr, nullptr, scores_bf, 0);
  softmax_bf16_kernel<<<dim3(B), dim3(256), 0, stream>>>(scores_bf, N);

  // attended = 0.1 * (a @ v) + q -> f32 ; layernorm -> bf16
  gemm(scores_bf, v_bf, B, M, M, N, 0.1f, nullptr, q_f32, att_f32, nullptr, nullptr, 0);
  layernorm_kernel<<<dim3(B), dim3(256), 0, stream>>>(att_f32, ln_gamma, ln_beta, normed_bf, M);

  // fc_hash = normed @ W_hash_pad + b_hash ; code/prob fused -> d_out
  gemm(normed_bf, Wh_pad, B, CPAD, C, M, 1.f, b_hash, nullptr, out_code, out_prob, nullptr, 1);

  // fc_cls = code @ W_cls + b_cls
  cls_kernel<<<dim3((unsigned)(((size_t)B * K + 255) / 256)), dim3(256), 0, stream>>>(
      out_code, W_cls, b_cls, out_cls, B, C, K);
}
